// FluxlikeResnetBlock_64003602645713
// MI455X (gfx1250) — matrix operand order repaired and checked by reading, not compiled, not run on hardware
//
#include <hip/hip_runtime.h>
#include <hip/hip_bf16.h>
#include <stdint.h>

typedef __attribute__((ext_vector_type(16))) __bf16 v16bf;
typedef __attribute__((ext_vector_type(8)))  __bf16 v8bf;
typedef __attribute__((ext_vector_type(8)))  float  v8f;

#define DEV __device__ __forceinline__

// ---------- bf16 helpers (RNE convert via bit ops) ----------
DEV __bf16 f2bf(float f) {
  unsigned u; __builtin_memcpy(&u, &f, 4);
  unsigned r = u + 0x7FFFu + ((u >> 16) & 1u);
  unsigned short h = (unsigned short)(r >> 16);
  __bf16 o; __builtin_memcpy(&o, &h, 2); return o;
}
DEV float bf2f(__bf16 b) {
  unsigned short h; __builtin_memcpy(&h, &b, 2);
  unsigned u = ((unsigned)h) << 16;
  float f; __builtin_memcpy(&f, &u, 4); return f;
}

// ---------- CDNA5 async copy: global -> LDS (ASYNCcnt tracked) ----------
DEV void async_b128(const void* ldsp, const void* gsrc) {
  unsigned lo = (unsigned)(uintptr_t)ldsp;   // low 32 bits of LDS-aperture addr == LDS byte offset
  asm volatile("global_load_async_to_lds_b128 %0, %1, off"
               :: "v"(lo), "v"(gsrc) : "memory");
}
DEV void wait_async_le8() { asm volatile("s_wait_asynccnt 0x8" ::: "memory"); }
DEV void wait_async_le2() { asm volatile("s_wait_asynccnt 0x2" ::: "memory"); }
DEV void wait_async_0()   { asm volatile("s_wait_asynccnt 0x0" ::: "memory"); }

// ---------- WMMA fragment loaders ----------
// A-matrix 16x32 bf16: lane%16 = row; lanes 0-15: K=0..15, lanes 16-31: K=16..31.
DEV v16bf load_afrag(const __bf16* rowk, int lane) {
  return *(const v16bf*)(rowk + (lane >> 4) * 16);
}
// B-matrix 32x16 bf16: lane%16 = column; lanes 0-15: K=0..15, lanes 16-31: K=16..31.
DEV v16bf load_bfrag(const __bf16* colk, int lane) {
  return *(const v16bf*)(colk + (lane >> 4) * 16);
}
DEV v8f wmma_bf16(v16bf a, v16bf b, v8f c) {
  return __builtin_amdgcn_wmma_f32_16x16x32_bf16(false, a, false, b, (short)0, c, false, false);
}

// ---------- problem constants ----------
// B=8 C=512 H=W=32 HW=1024 T=8192 S=77 EMB=768 NH=8 HD=64 L=1101 Lp=1120 Mc=624

// ---------- small elementwise kernels ----------
__global__ void f2bf_kernel(const float* __restrict__ in, __bf16* __restrict__ dst, size_t n) {
  size_t i = (size_t)blockIdx.x * 256 + threadIdx.x;
  if (i < n) dst[i] = f2bf(in[i]);
}

__global__ void cond_pad_kernel(const float* __restrict__ cond, __bf16* __restrict__ dst) {
  size_t i = (size_t)blockIdx.x * 256 + threadIdx.x;   // 624*768 total
  if (i >= (size_t)624 * 768) return;
  size_t r = i / 768;
  dst[i] = (r < 616) ? f2bf(cond[i]) : f2bf(0.f);
}

// params[b, 6C] = silu(final_emb[b]) @ mod_w^T + mod_b
__global__ void mod_kernel(const float* __restrict__ emb, const float* __restrict__ w,
                           const float* __restrict__ bias, float* __restrict__ params) {
  int j = blockIdx.x * 256 + threadIdx.x;              // 8*3072
  if (j >= 8 * 3072) return;
  int b = j / 3072, o = j % 3072;
  const float* e = emb + (size_t)b * 768;
  const float* wr = w + (size_t)o * 768;
  float acc = bias[o];
  for (int k = 0; k < 768; ++k) {
    float xv = e[k];
    acc += (xv / (1.f + __expf(-xv))) * wr[k];
  }
  params[j] = acc;
}

// ---------- RMS over channels + modulate -> bf16 token-major activations ----------
__global__ __launch_bounds__(256)
void norm_mod_kernel(const float* __restrict__ src,   // [B,C,HW]
                     const float* __restrict__ scale, // [C]
                     const float* __restrict__ params, int sh_off, int sc_off,
                     __bf16* __restrict__ outh)       // [T, C]
{
  __shared__ __bf16 tile[512 * 33];
  __shared__ float red[256];
  __shared__ float rr[32];
  const int b  = blockIdx.x >> 5;
  const int hr = blockIdx.x & 31;
  const int w  = threadIdx.x & 31;
  const int cg = threadIdx.x >> 5;
  const float* sp = src + (size_t)b * 512 * 1024 + hr * 32;
  float ss = 0.f;
  for (int cc = 0; cc < 64; ++cc) {
    int c = cg * 64 + cc;
    float v = sp[(size_t)c * 1024 + w];     // coalesced over w
    tile[c * 33 + w] = f2bf(v);
    ss += v * v;
  }
  red[threadIdx.x] = ss;
  __syncthreads();
  if (cg == 0) {
    float t = 0.f;
#pragma unroll
    for (int g = 0; g < 8; ++g) t += red[g * 32 + w];
    rr[w] = rsqrtf(t * (1.f / 512.f) + 1e-6f);
  }
  __syncthreads();
  const float* psc = params + b * 3072 + sc_off;
  const float* psh = params + b * 3072 + sh_off;
#pragma unroll
  for (int i = 0; i < 4; ++i) {
    int w2 = cg * 4 + i;
    float rv = rr[w2];
    size_t t = (size_t)b * 1024 + hr * 32 + w2;
    for (int c0 = w; c0 < 512; c0 += 32) {
      float v = bf2f(tile[c0 * 33 + w2]);
      float o = v * rv * scale[c0] * (1.f + psc[c0]) + psh[c0];
      outh[t * 512 + c0] = f2bf(o);       // coalesced over c0
    }
  }
}

// ---------- WMMA GEMM with double-buffered async LDS staging ----------
// C[M,N] = A[M,K](bf16) * W[N,K]^T(bf16); block = 128 threads = 4 waves,
// block tile = 64(M) x 64(N), K-step 64 (8 WMMAs per sync round).
// A/B tiles staged via GLOBAL_LOAD_ASYNC_TO_LDS_B128 (ASYNCcnt), double buffered.
// EPI 0: store f32. EPI 1: +bias, tanh-GELU, store bf16.
// EPI 2: +bias, then out[b,n,p] = xres[b,n,p] + gate[b,goff+n]*val (M=b*1024+p, N=512)
template<int EPI>
__global__ __launch_bounds__(128)
void gemm_bf16_kernel(const __bf16* __restrict__ A, const __bf16* __restrict__ Wt,
                      int M, int N, int K,
                      float* __restrict__ outF, __bf16* __restrict__ outB,
                      const float* __restrict__ bias,
                      const float* __restrict__ params, int gate_off,
                      const float* __restrict__ xres)
{
  __shared__ __bf16 lA[2][64 * 64];     // 8KB per buffer
  __shared__ __bf16 lB[2][64 * 64];
  const int tid  = threadIdx.x;
  const int lane = tid & 31;
  const int wid  = tid >> 5;
  const int m0b  = blockIdx.y * 64;
  const int n0   = blockIdx.x * 64;
  const int lcol = lane & 15;

  // staging: thread -> one 64B half-row of each tile (4 x B128 per matrix)
  const int r  = tid >> 1;              // row 0..63
  const int hf = tid & 1;               // 32-element half
  int arow = m0b + r; if (arow >= M) arow = M - 1;   // clamp (padded-M safety)
  const __bf16* aSrc = A  + (size_t)arow * K + hf * 32;
  const __bf16* bSrc = Wt + (size_t)(n0 + r) * K + hf * 32;
  __bf16* aDst0 = &lA[0][r * 64 + hf * 32];
  __bf16* bDst0 = &lB[0][r * 64 + hf * 32];
  __bf16* aDst1 = &lA[1][r * 64 + hf * 32];
  __bf16* bDst1 = &lB[1][r * 64 + hf * 32];

  const int nk = K / 64;
  // prologue: tile 0 -> buf 0
#pragma unroll
  for (int c = 0; c < 4; ++c) {
    async_b128(aDst0 + c * 8, aSrc + c * 8);
    async_b128(bDst0 + c * 8, bSrc + c * 8);
  }

  v8f z = {0.f, 0.f, 0.f, 0.f, 0.f, 0.f, 0.f, 0.f};
  v8f acc[4] = {z, z, z, z};
  for (int kt = 0; kt < nk; ++kt) {
    if (kt + 1 < nk) {
      const int k0 = (kt + 1) * 64;
      __bf16* ad = ((kt + 1) & 1) ? aDst1 : aDst0;
      __bf16* bd = ((kt + 1) & 1) ? bDst1 : bDst0;
#pragma unroll
      for (int c = 0; c < 4; ++c) {
        async_b128(ad + c * 8, aSrc + k0 + c * 8);
        async_b128(bd + c * 8, bSrc + k0 + c * 8);
      }
      wait_async_le8();                 // tile kt done (in-order), tile kt+1 in flight
    } else {
      wait_async_0();
    }
    __syncthreads();                    // all waves' async stores visible
    const int cb = kt & 1;
    const __bf16* la = &lA[cb][(wid * 16 + lcol) * 64];
    // hoist A and k 0..31 B fragments before the first WMMAs for LDS/WMMA overlap
    v16bf a0 = load_afrag(la, lane);
    v16bf a1 = load_afrag(la + 32, lane);
    v16bf b00 = load_bfrag(&lB[cb][(0 * 16 + lcol) * 64], lane);
    v16bf b10 = load_bfrag(&lB[cb][(1 * 16 + lcol) * 64], lane);
    v16bf b20 = load_bfrag(&lB[cb][(2 * 16 + lcol) * 64], lane);
    v16bf b30 = load_bfrag(&lB[cb][(3 * 16 + lcol) * 64], lane);
    acc[0] = wmma_bf16(a0, b00, acc[0]);
    acc[1] = wmma_bf16(a0, b10, acc[1]);
    acc[2] = wmma_bf16(a0, b20, acc[2]);
    acc[3] = wmma_bf16(a0, b30, acc[3]);
    v16bf b01 = load_bfrag(&lB[cb][(0 * 16 + lcol) * 64 + 32], lane);
    v16bf b11 = load_bfrag(&lB[cb][(1 * 16 + lcol) * 64 + 32], lane);
    v16bf b21 = load_bfrag(&lB[cb][(2 * 16 + lcol) * 64 + 32], lane);
    v16bf b31 = load_bfrag(&lB[cb][(3 * 16 + lcol) * 64 + 32], lane);
    acc[0] = wmma_bf16(a1, b01, acc[0]);
    acc[1] = wmma_bf16(a1, b11, acc[1]);
    acc[2] = wmma_bf16(a1, b21, acc[2]);
    acc[3] = wmma_bf16(a1, b31, acc[3]);
    __syncthreads();                    // done reading before buffer reuse
  }

  const int m0 = m0b + wid * 16;
  if (m0 >= M) return;                  // after all barriers: safe
#pragma unroll
  for (int nt = 0; nt < 4; ++nt) {
#pragma unroll
    for (int rr2 = 0; rr2 < 8; ++rr2) {
      int m = m0 + rr2 + (lane >> 4) * 8;
      int n = n0 + nt * 16 + lcol;
      float v = acc[nt][rr2];
      if (EPI == 0) {
        outF[(size_t)m * N + n] = v;
      } else if (EPI == 1) {
        v += bias[n];
        float g = 0.5f * v * (1.f + tanhf(0.7978845608f * (v + 0.044715f * v * v * v)));
        outB[(size_t)m * N + n] = f2bf(g);
      } else {
        v += bias[n];
        int bb = m >> 10, p = m & 1023;
        float gt = params[bb * 3072 + gate_off + n];
        size_t idx = ((size_t)bb * 512 + n) * 1024 + p;
        outF[idx] = xres[idx] + gt * v;
      }
    }
  }
}

// ---------- RoPE + QK-RMSNorm + pack Q/K bf16 [BH,Lp,64], V^T bf16 [BH,64,Lp] ----------
__global__ __launch_bounds__(256)
void qkv_prep_kernel(const float* __restrict__ qkvs,  // [T,1536]
                     const float* __restrict__ qkvc,  // [624,1536]
                     const float* __restrict__ rf,    // [32,2]
                     const float* __restrict__ qnw, const float* __restrict__ knw,
                     __bf16* __restrict__ qb, __bf16* __restrict__ kb,
                     __bf16* __restrict__ vt)
{
  const int lane = threadIdx.x & 31;
  const int wid  = threadIdx.x >> 5;
  const int R = blockIdx.x * 8 + wid;        // [0, 8*8*1120)
  const int b = R / (8 * 1120);
  const int rem = R % (8 * 1120);
  const int nh = rem / 1120;
  const int l  = rem % 1120;
  float q0 = 0.f, q1 = 0.f, k0 = 0.f, k1 = 0.f, v0 = 0.f, v1 = 0.f;
  if (l < 1024) {
    const float* base = qkvs + (size_t)(b * 1024 + l) * 1536 + nh * 64 + 2 * lane;
    float qe = base[0],    qo = base[1];
    float ke = base[512],  ko = base[513];
    v0 = base[1024]; v1 = base[1025];
    float gy = (float)(l >> 5), gx = (float)(l & 31);
    float ang = gy * rf[2 * lane] + gx * rf[2 * lane + 1];
    float cs = cosf(ang), sn = sinf(ang);
    q0 = qe * cs - qo * sn;  q1 = qo * cs + qe * sn;
    k0 = ke * cs - ko * sn;  k1 = ko * cs + ke * sn;
  } else if (l < 1101) {
    int sidx = l - 1024;
    const float* base = qkvc + (size_t)(b * 77 + sidx) * 1536 + nh * 64 + 2 * lane;
    float qe = base[0],    qo = base[1];
    float ke = base[512],  ko = base[513];
    v0 = base[1024]; v1 = base[1025];
    float invf = __powf(10000.f, -(float)(2 * lane) * (1.f / 64.f));
    float ang = (float)sidx * invf;
    float cs = cosf(ang), sn = sinf(ang);
    q0 = qe * cs - qo * sn;  q1 = qo * cs + qe * sn;
    k0 = ke * cs - ko * sn;  k1 = ko * cs + ke * sn;
  }
  float sq = q0 * q0 + q1 * q1;
  float sk = k0 * k0 + k1 * k1;
#pragma unroll
  for (int m = 1; m < 32; m <<= 1) { sq += __shfl_xor(sq, m, 32); sk += __shfl_xor(sk, m, 32); }
  float rq = rsqrtf(sq * (1.f / 64.f) + 1e-6f);
  float rk = rsqrtf(sk * (1.f / 64.f) + 1e-6f);
  q0 *= rq * qnw[2 * lane]; q1 *= rq * qnw[2 * lane + 1];
  k0 *= rk * knw[2 * lane]; k1 *= rk * knw[2 * lane + 1];
  size_t rowq = ((size_t)(b * 8 + nh) * 1120 + l) * 64 + 2 * lane;
  qb[rowq] = f2bf(q0); qb[rowq + 1] = f2bf(q1);
  kb[rowq] = f2bf(k0); kb[rowq + 1] = f2bf(k1);
  size_t vb = ((size_t)(b * 8 + nh) * 64 + 2 * lane) * 1120 + l;
  vt[vb] = f2bf(v0); vt[vb + 1120] = f2bf(v1);
}

// ---------- flash attention: block = 8 waves sharing one (b,nh); K/V tiles ----------
// async-staged to LDS once per block; per-wave 16-query tile, online softmax.
__global__ __launch_bounds__(256)
void attn_kernel(const __bf16* __restrict__ qb, const __bf16* __restrict__ kb,
                 const __bf16* __restrict__ vt, __bf16* __restrict__ obf)
{
  __shared__ __bf16 kT[2][32 * 64];           // 32 tokens x 64 hd
  __shared__ __bf16 vT[2][64 * 32];           // 64 hd x 32 tokens (V^T)
  __shared__ __bf16 plds[8][512];             // per-wave 16x32 bf16 P tile
  const int tid  = threadIdx.x;
  const int lane = tid & 31;
  const int wid  = tid >> 5;
  const int bh = blockIdx.x >> 3;             // b*8 + nh  (8 blocks per bh)
  const int qt = (blockIdx.x & 7) * 8 + wid;  // 0..63
  const __bf16* qp = qb + (size_t)bh * 1120 * 64;
  const __bf16* kp = kb + (size_t)bh * 1120 * 64;
  const __bf16* vp = vt + (size_t)bh * 64 * 1120;
  const int lcol = lane & 15;
  const int q0r = qt * 16;
  v16bf a0 = load_afrag(qp + (size_t)(q0r + lcol) * 64, lane);
  v16bf a1 = load_afrag(qp + (size_t)(q0r + lcol) * 64 + 32, lane);
  v8f z = {0.f, 0.f, 0.f, 0.f, 0.f, 0.f, 0.f, 0.f};
  v8f o0 = z, o1 = z, o2 = z, o3 = z;
  float rm[8], rl[8];
#pragma unroll
  for (int r = 0; r < 8; ++r) { rm[r] = -1e30f; rl[r] = 0.f; }
  __bf16* pl = plds[wid];

  // async staging assignment (per thread, both tiles, 16B each)
  const int rk = tid >> 3, pk = tid & 7;      // K tile: token row 0..31, 8x16B parts
  const int rv = tid >> 2, pv = tid & 3;      // V tile: hd row 0..63, 4x16B parts
  const __bf16* kSrc = kp + (size_t)rk * 64 + pk * 8;
  const __bf16* vSrc = vp + (size_t)rv * 1120 + pv * 8;
  // prologue: tile 0
  async_b128(&kT[0][rk * 64 + pk * 8], kSrc);
  async_b128(&vT[0][rv * 32 + pv * 8], vSrc);

  const int njt = 1120 / 32;                  // 35 (uniform across block)
  for (int jt = 0; jt < njt; ++jt) {
    const int j0 = jt * 32;
    if (jt + 1 < njt) {
      const int nb = (jt + 1) & 1;
      async_b128(&kT[nb][rk * 64 + pk * 8], kSrc + (size_t)(j0 + 32) * 64);
      async_b128(&vT[nb][rv * 32 + pv * 8], vSrc + (j0 + 32));
      wait_async_le2();
    } else {
      wait_async_0();
    }
    __syncthreads();
    const int cb = jt & 1;
    const __bf16* kt_ = kT[cb];
    const __bf16* vt_ = vT[cb];
    v16bf b00 = load_bfrag(kt_ + lcol * 64, lane);
    v16bf b01 = load_bfrag(kt_ + lcol * 64 + 32, lane);
    v16bf b10 = load_bfrag(kt_ + (16 + lcol) * 64, lane);
    v16bf b11 = load_bfrag(kt_ + (16 + lcol) * 64 + 32, lane);
    v8f s0 = wmma_bf16(a0, b00, z); s0 = wmma_bf16(a1, b01, s0);
    v8f s1 = wmma_bf16(a0, b10, z); s1 = wmma_bf16(a1, b11, s1);
    const bool ok0 = (j0 + lcol) < 1101;
    const bool ok1 = (j0 + 16 + lcol) < 1101;
#pragma unroll
    for (int r = 0; r < 8; ++r) {
      float v0 = ok0 ? s0[r] * 0.125f : -1e30f;   // 1/sqrt(64)
      float v1 = ok1 ? s1[r] * 0.125f : -1e30f;
      float tm = fmaxf(v0, v1);
#pragma unroll
      for (int m = 1; m < 16; m <<= 1) tm = fmaxf(tm, __shfl_xor(tm, m, 32));
      float nm = fmaxf(rm[r], tm);
      float al = __expf(rm[r] - nm);
      rm[r] = nm;
      float p0 = __expf(v0 - nm);
      float p1 = __expf(v1 - nm);
      float ts = p0 + p1;
#pragma unroll
      for (int m = 1; m < 16; m <<= 1) ts += __shfl_xor(ts, m, 32);
      rl[r] = rl[r] * al + ts;
      o0[r] *= al; o1[r] *= al; o2[r] *= al; o3[r] *= al;
      int row = r + (lane >> 4) * 8;
      pl[row * 32 + lcol]      = f2bf(p0);
      pl[row * 32 + 16 + lcol] = f2bf(p1);
    }
    asm volatile("s_wait_dscnt 0x0" ::: "memory");   // wave-internal LDS RAW fence
    v16bf pa  = load_afrag(pl + lcol * 32, lane);
    v16bf vb0 = load_bfrag(vt_ + (0 * 16 + lcol) * 32, lane);
    v16bf vb1 = load_bfrag(vt_ + (1 * 16 + lcol) * 32, lane);
    v16bf vb2 = load_bfrag(vt_ + (2 * 16 + lcol) * 32, lane);
    v16bf vb3 = load_bfrag(vt_ + (3 * 16 + lcol) * 32, lane);
    o0 = wmma_bf16(pa, vb0, o0);
    o1 = wmma_bf16(pa, vb1, o1);
    o2 = wmma_bf16(pa, vb2, o2);
    o3 = wmma_bf16(pa, vb3, o3);
    __syncthreads();                    // before buffer reuse
  }
  const int b = bh >> 3, nh = bh & 7;
#pragma unroll
  for (int r = 0; r < 8; ++r) {
    int row = r + (lane >> 4) * 8;
    size_t t = (size_t)b * 1024 + q0r + row;
    float inv = 1.f / rl[r];
    __bf16* op = obf + t * 512 + nh * 64 + lcol;
    op[0]  = f2bf(o0[r] * inv);
    op[16] = f2bf(o1[r] * inv);
    op[32] = f2bf(o2[r] * inv);
    op[48] = f2bf(o3[r] * inv);
  }
}

// ---------- host launch ----------
extern "C" void kernel_launch(void* const* d_in, const int* in_sizes, int n_in,
                              void* d_out, int out_size, void* d_ws, size_t ws_size,
                              hipStream_t stream) {
  (void)in_sizes; (void)n_in; (void)out_size; (void)ws_size;
  const float* x       = (const float*)d_in[0];
  const float* cond    = (const float*)d_in[1];
  const float* femb    = (const float*)d_in[2];
  const float* mod_w   = (const float*)d_in[3];
  const float* mod_b   = (const float*)d_in[4];
  const float* n1s     = (const float*)d_in[5];
  const float* n2s     = (const float*)d_in[6];
  const float* qkv_s_w = (const float*)d_in[7];
  const float* qkv_c_w = (const float*)d_in[8];
  const float* ropef   = (const float*)d_in[9];
  const float* qnq     = (const float*)d_in[10];
  const float* qnk     = (const float*)d_in[11];
  const float* out_w   = (const float*)d_in[12];
  const float* out_b   = (const float*)d_in[13];
  const float* m1w     = (const float*)d_in[14];
  const float* m1b     = (const float*)d_in[15];
  const float* m2w     = (const float*)d_in[16];
  const float* m2b     = (const float*)d_in[17];
  float* out = (float*)d_out;

  char* ws = (char*)d_ws;
  size_t off = 0;
  auto alloc = [&](size_t bytes) -> char* {
    char* p = ws + off; off += (bytes + 255) & ~(size_t)255; return p;
  };
  float*  params = (float*) alloc((size_t)8 * 3072 * 4);
  __bf16* h1     = (__bf16*)alloc((size_t)8192 * 512 * 2);
  __bf16* wqs    = (__bf16*)alloc((size_t)1536 * 512 * 2);
  __bf16* wqc    = (__bf16*)alloc((size_t)1536 * 768 * 2);
  __bf16* wout   = (__bf16*)alloc((size_t)512 * 512 * 2);
  __bf16* wm1    = (__bf16*)alloc((size_t)2048 * 512 * 2);
  __bf16* wm2    = (__bf16*)alloc((size_t)512 * 2048 * 2);
  __bf16* condb  = (__bf16*)alloc((size_t)624 * 768 * 2);
  float*  qkvs   = (float*) alloc((size_t)8192 * 1536 * 4);
  float*  qkvc   = (float*) alloc((size_t)624 * 1536 * 4);
  __bf16* qbf    = (__bf16*)alloc((size_t)64 * 1120 * 64 * 2);
  __bf16* kbf    = (__bf16*)alloc((size_t)64 * 1120 * 64 * 2);
  __bf16* vtb    = (__bf16*)alloc((size_t)64 * 64 * 1120 * 2);
  __bf16* obf    = (__bf16*)alloc((size_t)8192 * 512 * 2);
  float*  x2     = (float*) alloc((size_t)8 * 512 * 1024 * 4);
  __bf16* h2     = (__bf16*)alloc((size_t)8192 * 512 * 2);
  __bf16* hid    = (__bf16*)alloc((size_t)8192 * 2048 * 2);

  auto conv = [&](const float* s, __bf16* d, size_t n) {
    f2bf_kernel<<<(unsigned)((n + 255) / 256), 256, 0, stream>>>(s, d, n);
  };
  conv(qkv_s_w, wqs, (size_t)1536 * 512);
  conv(qkv_c_w, wqc, (size_t)1536 * 768);
  conv(out_w,   wout, (size_t)512 * 512);
  conv(m1w,     wm1, (size_t)2048 * 512);
  conv(m2w,     wm2, (size_t)512 * 2048);
  cond_pad_kernel<<<(624 * 768 + 255) / 256, 256, 0, stream>>>(cond, condb);
  mod_kernel<<<(8 * 3072 + 255) / 256, 256, 0, stream>>>(femb, mod_w, mod_b, params);

  // norm1 + modulate (sh1 @ 0, sc1 @ 512)
  norm_mod_kernel<<<256, 256, 0, stream>>>(x, n1s, params, 0, 512, h1);

  // qkv GEMMs (fp32 outputs for RoPE/QK-norm precision)
  {
    dim3 g(1536 / 64, 8192 / 64);
    gemm_bf16_kernel<0><<<g, 128, 0, stream>>>(h1, wqs, 8192, 1536, 512,
                                               qkvs, nullptr, nullptr, nullptr, 0, nullptr);
  }
  {
    dim3 g(1536 / 64, 10);   // ceil(624/64)
    gemm_bf16_kernel<0><<<g, 128, 0, stream>>>(condb, wqc, 624, 1536, 768,
                                               qkvc, nullptr, nullptr, nullptr, 0, nullptr);
  }

  // RoPE + QKNorm + pack
  qkv_prep_kernel<<<(8 * 8 * 1120) / 8, 256, 0, stream>>>(qkvs, qkvc, ropef, qnq, qnk,
                                                          qbf, kbf, vtb);
  // attention
  attn_kernel<<<(64 * 64) / 8, 256, 0, stream>>>(qbf, kbf, vtb, obf);

  // out projection + gated residual -> x2 (g1 @ 1024)
  {
    dim3 g(512 / 64, 8192 / 64);
    gemm_bf16_kernel<2><<<g, 128, 0, stream>>>(obf, wout, 8192, 512, 512,
                                               x2, nullptr, out_b, params, 1024, x);
  }

  // norm2 + modulate (sh2 @ 1536, sc2 @ 2048)
  norm_mod_kernel<<<256, 256, 0, stream>>>(x2, n2s, params, 1536, 2048, h2);

  // MLP
  {
    dim3 g(2048 / 64, 8192 / 64);
    gemm_bf16_kernel<1><<<g, 128, 0, stream>>>(h2, wm1, 8192, 2048, 512,
                                               nullptr, hid, m1b, nullptr, 0, nullptr);
  }
  {
    dim3 g(512 / 64, 8192 / 64);
    gemm_bf16_kernel<2><<<g, 128, 0, stream>>>(hid, wm2, 8192, 512, 2048,
                                               out, nullptr, m2b, params, 2560, x2);
  }
}
